// ClauseDecoder_81690277970344
// MI455X (gfx1250) — matrix operand order repaired and checked by reading, not compiled, not run on hardware
//
#include <hip/hip_runtime.h>
#include <hip/hip_bf16.h>
#include <stdint.h>

// ---------------------------------------------------------------------------
// ClauseDecoder on MI455X (gfx1250): fused gather + 3-layer MLP.
//  - f16 WMMA (v_wmma_f32_16x16x32_f16) with f32 accumulation
//  - weights pre-transposed to [Nout][Kin] f16: B fragment = 2x16B/lane
//  - 64 clauses per block, 512 threads (16 wave32): each B fragment feeds
//    4 M-tiles -> halves L2 weight streaming vs 32-row blocks
//  - fully unrolled, software-pipelined K loop (fresh regs for next B)
//  - biases staged to LDS via global_load_async_to_lds_b128 (ASYNCcnt)
// ---------------------------------------------------------------------------

typedef __attribute__((ext_vector_type(16))) _Float16 v16h;
typedef __attribute__((ext_vector_type(8)))  float    v8f;

union FragU {
    v16h h;
    uint4 q[2];
};

// ---- workspace layout (f16 element offsets) -------------------------------
#define OFF_W1U 0                            // Wb1^T : [512][768]
#define OFF_W1B (OFF_W1U + 512 * 768)        // Wt1^T : [512][1024]
#define OFF_W2U (OFF_W1B + 512 * 1024)       // Wb2^T : [256][512]
#define OFF_W2B (OFF_W2U + 256 * 512)        // Wt2^T : [256][512]
#define OFF_WC1 (OFF_W2B + 256 * 512)        // Wc1^T : [256][256]
#define OFF_WC2 (OFF_WC1 + 256 * 256)        // Wc2   : [256]
#define WS_HALVES (OFF_WC2 + 256)

// ---------------------------------------------------------------------------
__global__ __launch_bounds__(256) void prep_weights(
    const float* __restrict__ Wb1, const float* __restrict__ Wt1,
    const float* __restrict__ Wb2, const float* __restrict__ Wt2,
    const float* __restrict__ Wc1, const float* __restrict__ Wc2,
    _Float16* __restrict__ ws)
{
    long t = (long)blockIdx.x * 256 + threadIdx.x;
    if (t >= WS_HALVES) return;
    if (t < OFF_W1B) {
        long i = t - OFF_W1U; long n = i / 768,  k = i % 768;
        ws[t] = (_Float16)Wb1[k * 512 + n];
    } else if (t < OFF_W2U) {
        long i = t - OFF_W1B; long n = i / 1024, k = i % 1024;
        ws[t] = (_Float16)Wt1[k * 512 + n];
    } else if (t < OFF_W2B) {
        long i = t - OFF_W2U; long n = i / 512,  k = i % 512;
        ws[t] = (_Float16)Wb2[k * 256 + n];
    } else if (t < OFF_WC1) {
        long i = t - OFF_W2B; long n = i / 512,  k = i % 512;
        ws[t] = (_Float16)Wt2[k * 256 + n];
    } else if (t < OFF_WC2) {
        long i = t - OFF_WC1; long n = i / 256,  k = i % 256;
        ws[t] = (_Float16)Wc1[k * 256 + n];
    } else {
        ws[t] = (_Float16)Wc2[t - OFF_WC2];
    }
}

// ---------------------------------------------------------------------------
// 64xNOUT GEMM layer: sOut = relu(sIn[64,KW] * WT^T + bias)
//   WT : [NOUT][KW] f16 row-major (pre-transposed); bias: f32 (in LDS)
//   16 waves; each wave owns TPW N-tiles x 4 M-tiles of 16x16.
//   K loop fully unrolled; next-step B fragments loaded into fresh
//   registers before the current WMMAs so loads overlap the matrix pipe.
// ---------------------------------------------------------------------------
template<int KW, int NOUT, int TPW>
__device__ inline void gemm64(const _Float16* __restrict__ sIn,
                              _Float16* __restrict__ sOut,
                              const _Float16* __restrict__ WT,
                              const float* __restrict__ bias,
                              int wv, int lane)
{
    constexpr int MT = 4;
    const int l  = lane & 15;
    const int hf = lane >> 4;

    v8f acc[MT][TPW];
#pragma unroll
    for (int m = 0; m < MT; ++m)
#pragma unroll
        for (int t = 0; t < TPW; ++t)
            acc[m][t] = (v8f){0.f, 0.f, 0.f, 0.f, 0.f, 0.f, 0.f, 0.f};

    auto loadA = [&](int mt, int kk) -> FragU {
        FragU a;
        const uint4* ap = reinterpret_cast<const uint4*>(
            sIn + (size_t)(mt * 16 + l) * KW + kk + hf * 8);
        a.q[0] = ap[0];   // K = kk + hf*8 + 0..7
        a.q[1] = ap[2];   // K = kk + 16 + hf*8 + 0..7
        return a;
    };
    auto loadB = [&](int t, int kk) -> FragU {
        FragU b;
        const int n0 = (wv * TPW + t) * 16;
        const uint4* bp = reinterpret_cast<const uint4*>(
            WT + (size_t)(n0 + l) * KW + kk + hf * 8);
        b.q[0] = bp[0];
        b.q[1] = bp[2];
        return b;
    };

    constexpr int KSTEPS = KW / 32;
    FragU bcur[TPW];
#pragma unroll
    for (int t = 0; t < TPW; ++t) bcur[t] = loadB(t, 0);

#pragma unroll
    for (int ks = 0; ks < KSTEPS; ++ks) {
        const int kk = ks * 32;
        FragU a[MT];
#pragma unroll
        for (int m = 0; m < MT; ++m) a[m] = loadA(m, kk);
        FragU bnxt[TPW];
        if (ks + 1 < KSTEPS) {
#pragma unroll
            for (int t = 0; t < TPW; ++t) bnxt[t] = loadB(t, kk + 32);
        }
#pragma unroll
        for (int t = 0; t < TPW; ++t)
#pragma unroll
            for (int m = 0; m < MT; ++m)
                acc[m][t] = __builtin_amdgcn_wmma_f32_16x16x32_f16(
                    false, a[m].h, false, bcur[t].h, (short)0,
                    acc[m][t], false, false);
        if (ks + 1 < KSTEPS) {
#pragma unroll
            for (int t = 0; t < TPW; ++t) bcur[t] = bnxt[t];
        }
    }

#pragma unroll
    for (int t = 0; t < TPW; ++t) {
        const int n0 = (wv * TPW + t) * 16;
        const float bv = bias[n0 + l];
#pragma unroll
        for (int m = 0; m < MT; ++m)
#pragma unroll
            for (int r = 0; r < 8; ++r) {
                float x = acc[m][t][r] + bv;
                x = x > 0.f ? x : 0.f;
                sOut[(size_t)(m * 16 + hf * 8 + r) * NOUT + n0 + l] =
                    (_Float16)x;
            }
    }
}

// ---------------------------------------------------------------------------
// Fused branch kernel: 64 clauses per block, 512 threads (16 wave32).
//   KIN = (ARITY+1)*256 : 768 (unary) / 1024 (binary)
// ---------------------------------------------------------------------------
template<int KIN, int ARITY>
__global__ __launch_bounds__(512) void mlp_branch(
    const float*     __restrict__ local_emb,   // [N,256] f32
    const float*     __restrict__ global_emb,  // [256]   f32
    const long long* __restrict__ idx,         // [M,ARITY] int64
    const _Float16*  __restrict__ W1T,         // [512][KIN]
    const float*     __restrict__ b1,          // [512]
    const _Float16*  __restrict__ W2T,         // [256][512]
    const float*     __restrict__ b2,          // [256]
    const _Float16*  __restrict__ Wc1T,        // [256][256]
    const float*     __restrict__ bc1,         // [256]
    const _Float16*  __restrict__ Wc2h,        // [256]
    const float*     __restrict__ bc2,         // [1]
    float*           __restrict__ out)         // [M]
{
    extern __shared__ __align__(16) unsigned char smemRaw[];
    _Float16* sX  = (_Float16*)smemRaw;        // [64][KIN]  (reused as h3)
    _Float16* sH1 = sX  + 64 * KIN;            // [64][512]
    _Float16* sH2 = sH1 + 64 * 512;            // [64][256]
    float*    sBias = (float*)(sH2 + 64 * 256);// [512 b1 | 256 b2 | 256 bc1]

    const int tid  = threadIdx.x;
    const int lane = tid & 31;
    const int wv   = tid >> 5;
    const long base = (long)blockIdx.x * 64;

    // ---- async-stage biases into LDS (overlaps with the gather below) -----
    if (tid < 256) {
        const float* src;
        uint32_t voff, ldsoff;
        const uint32_t biasLds = (uint32_t)(uintptr_t)(void*)sBias;
        if (wv < 4)      { src = b1;  voff = tid * 16;
                           ldsoff = biasLds + tid * 16; }
        else if (wv < 6) { src = b2;  voff = (tid - 128) * 16;
                           ldsoff = biasLds + 2048 + (tid - 128) * 16; }
        else             { src = bc1; voff = (tid - 192) * 16;
                           ldsoff = biasLds + 3072 + (tid - 192) * 16; }
        asm volatile("global_load_async_to_lds_b128 %0, %1, %2"
                     :: "v"(ldsoff), "v"(voff), "s"(src) : "memory");
    }

    // ---- gather: [64, KIN] = ARITY node rows + 1 global row, f32->f16 -----
    constexpr int NV4 = 64 * KIN / 4;
    for (int i4 = tid; i4 < NV4; i4 += 512) {
        const int c   = i4 / (KIN / 4);
        const int j   = (i4 - c * (KIN / 4)) * 4;
        const int seg = j >> 8;
        const int off = j & 255;
        float4 v;
        if (seg < ARITY) {
            const long long node = idx[(base + c) * ARITY + seg];
            v = *reinterpret_cast<const float4*>(
                    local_emb + (size_t)node * 256 + off);
        } else {
            v = *reinterpret_cast<const float4*>(global_emb + off);
        }
        union { _Float16 h[4]; uint2 u; } pk;
        pk.h[0] = (_Float16)v.x; pk.h[1] = (_Float16)v.y;
        pk.h[2] = (_Float16)v.z; pk.h[3] = (_Float16)v.w;
        *reinterpret_cast<uint2*>(sX + i4 * 4) = pk.u;
    }

    asm volatile("s_wait_asynccnt 0x0" ::: "memory");
    __syncthreads();

    const float* sB1  = sBias;
    const float* sB2  = sBias + 512;
    const float* sBc1 = sBias + 768;

    // ---- layer 1: [64,KIN] x [KIN,512] + b1, ReLU -------------------------
    gemm64<KIN, 512, 2>(sX, sH1, W1T, sB1, wv, lane);
    __syncthreads();

    // ---- layer 2: [64,512] x [512,256] + b2, common-layer ReLU ------------
    gemm64<512, 256, 1>(sH1, sH2, W2T, sB2, wv, lane);
    __syncthreads();

    // ---- common layer: [64,256] x [256,256] + bc1, ReLU (h3 -> sX) --------
    gemm64<256, 256, 1>(sH2, sX, Wc1T, sBc1, wv, lane);
    __syncthreads();

    // ---- final Linear(256,1): per-row dot ---------------------------------
    if (tid < 64) {
        const _Float16* row = sX + tid * 256;
        float s = bc2[0];
#pragma unroll 8
        for (int j = 0; j < 256; ++j)
            s += (float)row[j] * (float)Wc2h[j];
        out[base + tid] = s;
    }
}

// ---------------------------------------------------------------------------
extern "C" void kernel_launch(void* const* d_in, const int* in_sizes, int n_in,
                              void* d_out, int out_size, void* d_ws, size_t ws_size,
                              hipStream_t stream)
{
    (void)in_sizes; (void)n_in; (void)out_size; (void)ws_size;

    const float*     local_emb  = (const float*)d_in[0];
    const float*     global_emb = (const float*)d_in[1];
    const long long* unary_idx  = (const long long*)d_in[2];
    const long long* binary_idx = (const long long*)d_in[3];
    const float* Wb1 = (const float*)d_in[4];  const float* bb1 = (const float*)d_in[5];
    const float* Wb2 = (const float*)d_in[6];  const float* bb2 = (const float*)d_in[7];
    const float* Wt1 = (const float*)d_in[8];  const float* bt1 = (const float*)d_in[9];
    const float* Wt2 = (const float*)d_in[10]; const float* bt2 = (const float*)d_in[11];
    const float* Wc1 = (const float*)d_in[12]; const float* bc1 = (const float*)d_in[13];
    const float* Wc2 = (const float*)d_in[14]; const float* bc2 = (const float*)d_in[15];

    float* out = (float*)d_out;
    _Float16* ws = (_Float16*)d_ws;

    // 1) weights -> f16, transposed [Nout][Kin]
    {
        const int blocks = (WS_HALVES + 255) / 256;
        prep_weights<<<blocks, 256, 0, stream>>>(Wb1, Wt1, Wb2, Wt2, Wc1, Wc2, ws);
    }

    // dynamic LDS: (64*KIN + 64*512 + 64*256) halves + 1024 f32 of biases
    const size_t smemU = (size_t)(64 * 768  + 64 * 512 + 64 * 256) * 2 + 4096;
    const size_t smemB = (size_t)(64 * 1024 + 64 * 512 + 64 * 256) * 2 + 4096;
    (void)hipFuncSetAttribute((const void*)mlp_branch<768, 2>,
                              hipFuncAttributeMaxDynamicSharedMemorySize,
                              (int)smemU);
    (void)hipFuncSetAttribute((const void*)mlp_branch<1024, 3>,
                              hipFuncAttributeMaxDynamicSharedMemorySize,
                              (int)smemB);

    // 2) unary branch: 65536 clauses, 64 per block
    mlp_branch<768, 2><<<65536 / 64, 512, smemU, stream>>>(
        local_emb, global_emb, unary_idx,
        ws + OFF_W1U, bb1, ws + OFF_W2U, bb2,
        ws + OFF_WC1, bc1, ws + OFF_WC2, bc2,
        out);

    // 3) binary branch: 65536 clauses, 64 per block
    mlp_branch<1024, 3><<<65536 / 64, 512, smemB, stream>>>(
        local_emb, global_emb, binary_idx,
        ws + OFF_W1B, bt1, ws + OFF_W2B, bt2,
        ws + OFF_WC1, bc1, ws + OFF_WC2, bc2,
        out + 65536);
}
